// Multihead_Attention_F_Branch_real_1803886264332
// MI455X (gfx1250) — matrix operand order repaired and checked by reading, not compiled, not run on hardware
//
#include <hip/hip_runtime.h>

// MI455X / gfx1250 — wave32, WMMA 16x16x32 f16->f32, TDM async tensor load
typedef __attribute__((ext_vector_type(16))) _Float16 v16h;
typedef __attribute__((ext_vector_type(8)))  _Float16 v8h;
typedef __attribute__((ext_vector_type(8)))  float    v8f;
typedef unsigned int u32x4 __attribute__((ext_vector_type(4)));
typedef int          i32x8 __attribute__((ext_vector_type(8)));
typedef int          i32x4 __attribute__((ext_vector_type(4)));

#define N_  8
#define C_  64
#define F_  256
#define T_  256
#define H_  64

static __device__ inline v8f wmma_f16(v16h a, v16h b, v8f c) {
  return __builtin_amdgcn_wmma_f32_16x16x32_f16(false, a, false, b, (short)0, c, false, false);
}

// A fragment (16x32, f16): lane l holds row m = l&15; element e holds
// k = (e&7) | ((e&8)<<1) | ((l>>4)<<3)   (per CDNA5 ISA 16-bit A layout)
static __device__ inline v16h load_a_frag(const _Float16* base, int row0, int ld, int k0, int lane) {
  v16h r;
  const _Float16* p = base + (row0 + (lane & 15)) * ld + k0 + ((lane >> 4) << 3);
#pragma unroll
  for (int e = 0; e < 16; ++e) r[e] = p[(e & 7) + ((e & 8) << 1)];
  return r;
}

// B fragment (32x16, f16), operand stored N-major as X[n][k] (B = X^T).
// lane l holds col n = l&15; element e holds k = k0 + (e&7) + ((e&8)<<1) + (l>>4)*8 (matches A layout).
static __device__ inline v16h load_b_frag_nmajor(const _Float16* base, int col0, int ld, int k0, int lane) {
  v16h r;
  const _Float16* p = base + (col0 + (lane & 15)) * ld + k0 + ((lane >> 4) << 3);
#pragma unroll
  for (int e = 0; e < 16; ++e) r[e] = p[(e & 7) + ((e & 8) << 1)];
  return r;
}

// reduce across the 16 lanes sharing one C/D-matrix row (xor masks stay in half-wave)
static __device__ inline float rowred_sum(float x) {
#pragma unroll
  for (int m = 1; m < 16; m <<= 1) x += __shfl_xor(x, m, 32);
  return x;
}
static __device__ inline float rowred_max(float x) {
#pragma unroll
  for (int m = 1; m < 16; m <<= 1) x = fmaxf(x, __shfl_xor(x, m, 32));
  return x;
}

// LDS layout (dynamic):
//  [  0K.. 32K) w16 : Wq|Wk|Wv|Wt as f16, each 64x64
//  [ 32K.. 64K) hb  : 256x64 f16  (LN1 output; later reused for LN2(O))
//  [ 64K.. 96K) qb  : 256x64 f16
//  [ 96K..128K) kb  : 256x64 f16
//  [128K..160K) vb  : V TRANSPOSED: 64(h) x 256(f2) f16
//  [160K..288K) ab  : 256x256 f16 (attention probs)
//               ab's first 64K doubles as: xs (f32 TDM staging, stages 0-1)
//                                          outf (f32 result, stages 5-6)
//  [288K..+2.5K) par: g1|b1|g2|b2|g3|b3|bq|bk|bv|bt  (10 x 64 f32)
#define LDS_BYTES (288 * 1024 + 2560)

__global__ __launch_bounds__(256)
void mha_f_branch_kernel(const float* __restrict__ src,   // staged: tmp[b][f][c] ; direct: x
                         const float* __restrict__ x,     // original input (residual, direct mode)
                         float* __restrict__ dst,         // staged: tmp (in place) ; direct: out
                         const int staged,
                         const float* __restrict__ Wq, const float* __restrict__ bq,
                         const float* __restrict__ Wk, const float* __restrict__ bk,
                         const float* __restrict__ Wv, const float* __restrict__ bv,
                         const float* __restrict__ g1, const float* __restrict__ b1,
                         const float* __restrict__ g2, const float* __restrict__ b2,
                         const float* __restrict__ Wt, const float* __restrict__ bt,
                         const float* __restrict__ g3, const float* __restrict__ b3,
                         const float* __restrict__ alpha) {
  extern __shared__ char smem[];
  _Float16* w16 = (_Float16*)(smem);
  _Float16* hb  = (_Float16*)(smem + 32 * 1024);
  _Float16* qb  = (_Float16*)(smem + 64 * 1024);
  _Float16* kb  = (_Float16*)(smem + 96 * 1024);
  _Float16* vb  = (_Float16*)(smem + 128 * 1024);   // transposed: vb[h*256 + f2]
  _Float16* ab  = (_Float16*)(smem + 160 * 1024);
  float*    xs   = (float*)(smem + 160 * 1024);     // TDM staging (stages 0-1 only)
  float*    outf = (float*)(smem + 160 * 1024);     // f32 result (stages 5-6)
  float*    par  = (float*)(smem + 288 * 1024);

  const int tid  = threadIdx.x;
  const int lane = tid & 31;
  const int wid  = tid >> 5;
  const int hi   = lane >> 4;
  const int b    = blockIdx.x;     // b = n*T + t
  const int n    = b >> 8;
  const int t    = b & 255;

  const size_t nbase = (size_t)n * ((size_t)C_ * F_ * T_);

  // ---- Stage 0a (staged): kick off TDM DMA of this block's 64KB input tile ----
#if __has_builtin(__builtin_amdgcn_tensor_load_to_lds)
  if (staged && wid == 0) {
    const unsigned long long ga =
        (unsigned long long)(uintptr_t)(src + (size_t)b * (F_ * C_));
    const unsigned int la = (unsigned int)(uintptr_t)xs;  // flat LDS addr low32 == LDS offset
    u32x4 g0;
    g0[0] = 1u;                                    // count=1, user descriptor
    g0[1] = la;                                    // lds_addr (bytes)
    g0[2] = (unsigned int)(ga & 0xFFFFFFFFu);      // global_addr[31:0]
    g0[3] = (unsigned int)((ga >> 32) & 0x01FFFFFFu) | (2u << 30);  // addr[56:32], type=2
    i32x8 g1d;
    g1d[0] = 0x00020000;   // workgroup_mask=0, data_size=2 (4B)
    g1d[1] = 0x40000000;   // tensor_dim0[15:0]=16384 at bits 79:48 -> w1[31:16]
    g1d[2] = 0x00010000;   // tensor_dim1[15:0]=1 at bits 111:80 -> w2[31:16]
    g1d[3] = 0x40000000;   // tile_dim0=16384 at bits 127:112 -> w3[31:16]
    g1d[4] = 0x00000001;   // tile_dim1=1 (bits 143:128)
    g1d[5] = 16384;        // tensor_dim0_stride low32 (bits 207:160)
    g1d[6] = 0;
    g1d[7] = 0;
    i32x4 g2d = {0, 0, 0, 0};
    i32x4 g3d = {0, 0, 0, 0};
    i32x8 g4d = {0, 0, 0, 0, 0, 0, 0, 0};
    __builtin_amdgcn_tensor_load_to_lds(g0, g1d, g2d, g3d, g4d, 0);
  }
#endif

  // ---- Stage 0b: weights (f32->f16) + parameter vectors -> LDS (overlaps DMA) ----
  for (int i = tid; i < 4096; i += 256) {
    w16[i]         = (_Float16)Wq[i];
    w16[4096 + i]  = (_Float16)Wk[i];
    w16[8192 + i]  = (_Float16)Wv[i];
    w16[12288 + i] = (_Float16)Wt[i];
  }
  if (tid < 64) {
    par[tid]       = g1[tid];  par[64 + tid]  = b1[tid];
    par[128 + tid] = g2[tid];  par[192 + tid] = b2[tid];
    par[256 + tid] = g3[tid];  par[320 + tid] = b3[tid];
    par[384 + tid] = bq[tid];  par[448 + tid] = bk[tid];
    par[512 + tid] = bv[tid];  par[576 + tid] = bt[tid];
  }
#if __has_builtin(__builtin_amdgcn_tensor_load_to_lds)
  if (staged && wid == 0) {
    __builtin_amdgcn_s_wait_tensorcnt(0);
  }
#endif
  __syncthreads();

  // ---- Stage 1: LN1 over C (one row f per thread) ----
  {
    const int f = tid;
    float vals[64];
    float s = 0.f, ss = 0.f;
    if (staged) {
      const float* px = xs + (size_t)f * C_;    // tile already in LDS via TDM
#pragma unroll
      for (int c = 0; c < 64; ++c) { float u = px[c]; vals[c] = u; s += u; ss += u * u; }
    } else {
      const float* px = src + nbase + (size_t)f * T_ + t;
#pragma unroll
      for (int c = 0; c < 64; ++c) { float u = px[(size_t)c * (F_ * T_)]; vals[c] = u; s += u; ss += u * u; }
    }
    float mean = s * (1.f / 64.f);
    float var  = ss * (1.f / 64.f) - mean * mean;
    float rs   = rsqrtf(var + 1e-5f);
    _Float16* hrow = hb + f * 64;
#pragma unroll
    for (int i = 0; i < 8; ++i) {
      v8h pk;
#pragma unroll
      for (int j = 0; j < 8; ++j) {
        int c = i * 8 + j;
        pk[j] = (_Float16)((vals[c] - mean) * rs * par[c] + par[64 + c]);
      }
      *(v8h*)(hrow + i * 8) = pk;   // ds_store_b128
    }
  }
  __syncthreads();

  const int mt0 = wid * 2;  // each wave owns 2 M-tiles (32 rows of F)

  // ---- Stage 2: Q,K,V = LN1(x) @ W^T + b  (WMMA). V stored transposed ----
#pragma unroll
  for (int mi = 0; mi < 2; ++mi) {
    const int mt = mt0 + mi;
    v16h a0 = load_a_frag(hb, mt * 16, 64, 0, lane);
    v16h a1 = load_a_frag(hb, mt * 16, 64, 32, lane);
#pragma unroll
    for (int which = 0; which < 3; ++which) {
      const _Float16* wsel = w16 + which * 4096;
      const float* bias = par + 384 + which * 64;   // bq|bk|bv
#pragma unroll
      for (int nt = 0; nt < 4; ++nt) {
        v8f acc = {};
        acc = wmma_f16(a0, load_b_frag_nmajor(wsel, nt * 16, 64, 0, lane), acc);
        acc = wmma_f16(a1, load_b_frag_nmajor(wsel, nt * 16, 64, 32, lane), acc);
        const int col = nt * 16 + (lane & 15);
        const float bb = bias[col];
        if (which == 2) {
          // V transposed: vb[col(h)][row(f2)] — 8 consecutive rows -> one b128 store
          v8h pk;
#pragma unroll
          for (int v = 0; v < 8; ++v) pk[v] = (_Float16)(acc[v] + bb);
          *(v8h*)(vb + col * 256 + mt * 16 + hi * 8) = pk;
        } else {
          _Float16* dstw = (which == 0) ? qb : kb;
#pragma unroll
          for (int v = 0; v < 8; ++v)
            dstw[(mt * 16 + v + hi * 8) * 64 + col] = (_Float16)(acc[v] + bb);
        }
      }
    }
  }
  __syncthreads();

  // ---- Stage 3: S = Q @ K^T * scale; row softmax in registers; probs -> ab (f16) ----
  {
    const float scale = 0.125f;  // 1/sqrt(64)
#pragma unroll
    for (int mi = 0; mi < 2; ++mi) {
      const int mt = mt0 + mi;
      v16h a0 = load_a_frag(qb, mt * 16, 64, 0, lane);
      v16h a1 = load_a_frag(qb, mt * 16, 64, 32, lane);
      v8f s[16];
#pragma unroll
      for (int nt = 0; nt < 16; ++nt) {
        v8f acc = {};
        acc = wmma_f16(a0, load_b_frag_nmajor(kb, nt * 16, 64, 0, lane), acc);
        acc = wmma_f16(a1, load_b_frag_nmajor(kb, nt * 16, 64, 32, lane), acc);
        s[nt] = acc;
      }
#pragma unroll
      for (int v = 0; v < 8; ++v) {
        float mx = -3.0e38f;
#pragma unroll
        for (int nt = 0; nt < 16; ++nt) mx = fmaxf(mx, s[nt][v]);
        mx = rowred_max(mx);
        float p[16];
        float sum = 0.f;
#pragma unroll
        for (int nt = 0; nt < 16; ++nt) {
          p[nt] = __expf((s[nt][v] - mx) * scale);
          sum += p[nt];
        }
        sum = rowred_sum(sum);
        const float inv = 1.f / sum;
        const int row = mt * 16 + v + hi * 8;
#pragma unroll
        for (int nt = 0; nt < 16; ++nt)
          ab[row * 256 + nt * 16 + (lane & 15)] = (_Float16)(p[nt] * inv);
      }
    }
  }
  __syncthreads();

  // ---- Stage 4: O = A @ V ; LN2 over H fused in registers; -> hb (reused) ----
#pragma unroll
  for (int mi = 0; mi < 2; ++mi) {
    const int mt = mt0 + mi;
    v16h af[8];
#pragma unroll
    for (int ks = 0; ks < 8; ++ks) af[ks] = load_a_frag(ab, mt * 16, 256, ks * 32, lane);
    v8f o[4];
#pragma unroll
    for (int nt = 0; nt < 4; ++nt) {
      v8f acc = {};
#pragma unroll
      for (int ks = 0; ks < 8; ++ks)
        acc = wmma_f16(af[ks], load_b_frag_nmajor(vb, nt * 16, 256, ks * 32, lane), acc);
      o[nt] = acc;
    }
#pragma unroll
    for (int v = 0; v < 8; ++v) {
      float s = 0.f, ss = 0.f;
#pragma unroll
      for (int nt = 0; nt < 4; ++nt) { float u = o[nt][v]; s += u; ss += u * u; }
      s = rowred_sum(s); ss = rowred_sum(ss);
      float mean = s * (1.f / 64.f);
      float var  = ss * (1.f / 64.f) - mean * mean;
      float rs   = rsqrtf(var + 1e-5f);
      const int row = mt * 16 + v + hi * 8;
#pragma unroll
      for (int nt = 0; nt < 4; ++nt) {
        const int col = nt * 16 + (lane & 15);
        hb[row * 64 + col] = (_Float16)((o[nt][v] - mean) * rs * par[128 + col] + par[192 + col]);
      }
    }
  }
  __syncthreads();

  // ---- Stage 5: out = LN2(O) @ Wt^T + bt -> outf (f32, reuses ab region) ----
#pragma unroll
  for (int mi = 0; mi < 2; ++mi) {
    const int mt = mt0 + mi;
    v16h a0 = load_a_frag(hb, mt * 16, 64, 0, lane);
    v16h a1 = load_a_frag(hb, mt * 16, 64, 32, lane);
#pragma unroll
    for (int nt = 0; nt < 4; ++nt) {
      v8f acc = {};
      acc = wmma_f16(a0, load_b_frag_nmajor(w16 + 12288, nt * 16, 64, 0, lane), acc);
      acc = wmma_f16(a1, load_b_frag_nmajor(w16 + 12288, nt * 16, 64, 32, lane), acc);
      const int col = nt * 16 + (lane & 15);
      const float bb = par[576 + col];
#pragma unroll
      for (int v = 0; v < 8; ++v)
        outf[(mt * 16 + v + hi * 8) * 64 + col] = acc[v] + bb;
    }
  }
  __syncthreads();

  // ---- Stage 6: LN3 over C + PReLU; store (residual only in direct mode) ----
  {
    const int f = tid;
    const float* prow = outf + f * 64;
    float s = 0.f, ss = 0.f;
#pragma unroll
    for (int c = 0; c < 64; ++c) { float u = prow[c]; s += u; ss += u * u; }
    float mean = s * (1.f / 64.f);
    float var  = ss * (1.f / 64.f) - mean * mean;
    float rs   = rsqrtf(var + 1e-5f);
    const float al = alpha[0];
    if (staged) {
      float* po = dst + (size_t)b * (F_ * C_) + (size_t)f * C_;  // coalesced, residual in pass 3
#pragma unroll
      for (int c = 0; c < 64; ++c) {
        float u = (prow[c] - mean) * rs * par[256 + c] + par[320 + c];
        po[c] = (u >= 0.f) ? u : al * u;
      }
    } else {
      const float* px = x + nbase + (size_t)f * T_ + t;
      float*       po = dst + nbase + (size_t)f * T_ + t;
#pragma unroll
      for (int c = 0; c < 64; ++c) {
        float u = (prow[c] - mean) * rs * par[256 + c] + par[320 + c];
        u = (u >= 0.f) ? u : al * u;
        po[(size_t)c * (F_ * T_)] = u + px[(size_t)c * (F_ * T_)];
      }
    }
  }
}

// ---- Pass 1: x[N,C,F,T] -> tmp[b=n*T+t][f][c], coalesced both sides via LDS tile ----
__global__ __launch_bounds__(256)
void transpose_in_kernel(const float* __restrict__ x, float* __restrict__ tmp) {
  __shared__ float tile[32][33];
  const int nf = blockIdx.x;          // n*F + f
  const int n = nf >> 8, f = nf & 255;
  const int c0 = blockIdx.y * 32, t0 = blockIdx.z * 32;
  const int j = threadIdx.x & 31;
  const int i0 = threadIdx.x >> 5;    // 0..7
  const float* px = x + ((size_t)(n * C_ + c0)) * (F_ * T_) + (size_t)f * T_ + t0;
  __builtin_prefetch(px, 0, 1);
#pragma unroll
  for (int k = 0; k < 4; ++k) {
    int i = i0 + k * 8;               // c within tile
    tile[i][j] = px[(size_t)i * (F_ * T_) + j];   // coalesced along t
  }
  __syncthreads();
  float* pt = tmp + ((size_t)(n * T_ + t0)) * (F_ * C_) + (size_t)f * C_ + c0;
#pragma unroll
  for (int k = 0; k < 4; ++k) {
    int ti = i0 + k * 8;              // t within tile
    pt[(size_t)ti * (F_ * C_) + j] = tile[j][ti]; // coalesced along c
  }
}

// ---- Pass 3: out[n][c][f][t] = tmp[b][f][c] + x[n][c][f][t] ----
__global__ __launch_bounds__(256)
void transpose_out_kernel(const float* __restrict__ tmp, const float* __restrict__ x,
                          float* __restrict__ out) {
  __shared__ float tile[32][33];
  const int nf = blockIdx.x;
  const int n = nf >> 8, f = nf & 255;
  const int c0 = blockIdx.y * 32, t0 = blockIdx.z * 32;
  const int j = threadIdx.x & 31;
  const int i0 = threadIdx.x >> 5;
  const float* pt = tmp + ((size_t)(n * T_ + t0)) * (F_ * C_) + (size_t)f * C_ + c0;
  __builtin_prefetch(pt, 0, 1);
#pragma unroll
  for (int k = 0; k < 4; ++k) {
    int ti = i0 + k * 8;              // t within tile
    tile[ti][j] = pt[(size_t)ti * (F_ * C_) + j]; // coalesced along c
  }
  __syncthreads();
  const size_t gbase = ((size_t)(n * C_ + c0)) * (F_ * T_) + (size_t)f * T_ + t0;
  const float* px = x + gbase;
  float* po = out + gbase;
#pragma unroll
  for (int k = 0; k < 4; ++k) {
    int i = i0 + k * 8;               // c within tile
    po[(size_t)i * (F_ * T_) + j] = tile[j][i] + px[(size_t)i * (F_ * T_) + j];  // coalesced along t
  }
}

extern "C" void kernel_launch(void* const* d_in, const int* in_sizes, int n_in,
                              void* d_out, int out_size, void* d_ws, size_t ws_size,
                              hipStream_t stream) {
  const float* x     = (const float*)d_in[0];
  const float* Wq    = (const float*)d_in[1];
  const float* bq    = (const float*)d_in[2];
  const float* Wk    = (const float*)d_in[3];
  const float* bk    = (const float*)d_in[4];
  const float* Wv    = (const float*)d_in[5];
  const float* bv    = (const float*)d_in[6];
  const float* g1    = (const float*)d_in[7];
  const float* b1    = (const float*)d_in[8];
  const float* g2    = (const float*)d_in[9];
  const float* b2    = (const float*)d_in[10];
  const float* Wt    = (const float*)d_in[11];
  const float* bt    = (const float*)d_in[12];
  const float* g3    = (const float*)d_in[13];
  const float* b3    = (const float*)d_in[14];
  const float* alpha = (const float*)d_in[15];
  float* outp = (float*)d_out;

  const size_t elems = (size_t)N_ * C_ * F_ * T_;        // 33,554,432
  const bool staged = ws_size >= elems * sizeof(float);  // need 128 MB scratch
  float* tmp = (float*)d_ws;

  dim3 block(256);
  if (staged) {
    dim3 tgrid(N_ * F_, C_ / 32, T_ / 32);  // (2048, 2, 8)
    hipLaunchKernelGGL(transpose_in_kernel, tgrid, block, 0, stream, x, tmp);
    hipLaunchKernelGGL(mha_f_branch_kernel, dim3(N_ * T_), block, LDS_BYTES, stream,
                       tmp, x, tmp, 1,
                       Wq, bq, Wk, bk, Wv, bv, g1, b1, g2, b2, Wt, bt, g3, b3, alpha);
    hipLaunchKernelGGL(transpose_out_kernel, tgrid, block, 0, stream, tmp, x, outp);
  } else {
    hipLaunchKernelGGL(mha_f_branch_kernel, dim3(N_ * T_), block, LDS_BYTES, stream,
                       x, x, outp, 0,
                       Wq, bq, Wk, bk, Wv, bv, g1, b1, g2, b2, Wt, bt, g3, b3, alpha);
  }
}
